// HyperFusedMoE_65180423685435
// MI455X (gfx1250) — matrix operand order repaired and checked by reading, not compiled, not run on hardware
//
#include <hip/hip_runtime.h>
#include <hip/hip_bf16.h>

#define NB   16          // batch
#define CIN  256
#define COUT 256
#define HH   64
#define WW   64
#define HWP  4096        // H*W
#define NE   8           // experts
#define KT   2           // top_k
#define NG   8           // groups
#define CG   32          // channels per group
#define KDIM 288         // CG * 9 taps

typedef unsigned short ushort_t;
typedef __attribute__((ext_vector_type(16))) __bf16 v16bf;
typedef __attribute__((ext_vector_type(8)))  float  v8f;

union Frag { uint4 q[2]; v16bf v; };

static __device__ __forceinline__ ushort_t f2bf(float f) {
    unsigned int u = __float_as_uint(f);
    u += 0x7FFFu + ((u >> 16) & 1u);   // round-to-nearest-even
    return (ushort_t)(u >> 16);
}
static __device__ __forceinline__ float silu(float x) {
    return x / (1.f + __expf(-x));
}

// ---------------- kernel 1: per-channel mean / unbiased std ----------------
__global__ void k_chanstats(const float* __restrict__ x,
                            float* __restrict__ mean, float* __restrict__ stdv) {
    int bc = blockIdx.x;                       // [0, NB*CIN)
    const float* p = x + (size_t)bc * HWP;
    float s = 0.f, q = 0.f;
    for (int i = threadIdx.x; i < HWP; i += 256) { float v = p[i]; s += v; q += v * v; }
    for (int o = 16; o > 0; o >>= 1) { s += __shfl_xor(s, o, 32); q += __shfl_xor(q, o, 32); }
    __shared__ float ls[8], lq[8];
    int lane = threadIdx.x & 31, w = threadIdx.x >> 5;
    if (lane == 0) { ls[w] = s; lq[w] = q; }
    __syncthreads();
    if (threadIdx.x == 0) {
        s = 0.f; q = 0.f;
        for (int i = 0; i < 8; ++i) { s += ls[i]; q += lq[i]; }
        float m = s / (float)HWP;
        float var = (q - (float)HWP * m * m) / (float)(HWP - 1);   // ddof=1
        mean[bc] = m;
        stdv[bc] = sqrtf(fmaxf(var, 0.f));
    }
}

// ---------------- kernel 2: router (one warp per batch sample) ----------------
__global__ void k_router(const float* __restrict__ mean, const float* __restrict__ stdv,
                         const float* __restrict__ rwm,
                         int* __restrict__ idx, float* __restrict__ rwt) {
    int b = threadIdx.x >> 5, lane = threadIdx.x & 31;
    if (b >= NB) return;
    float logit[NE];
    for (int e = 0; e < NE; ++e) {
        float acc = 0.f;
        for (int c = lane; c < 2 * CIN; c += 32) {
            float sv = (c < CIN) ? mean[b * CIN + c] : stdv[b * CIN + c - CIN];
            acc += sv * rwm[e * 2 * CIN + c];
        }
        for (int o = 16; o > 0; o >>= 1) acc += __shfl_xor(acc, o, 32);
        logit[e] = acc;
    }
    // softmax #1 (Sequential's softmax)
    float mx = logit[0];
    for (int e = 1; e < NE; ++e) mx = fmaxf(mx, logit[e]);
    float p1[NE]; float sum = 0.f;
    for (int e = 0; e < NE; ++e) { p1[e] = __expf(logit[e] - mx); sum += p1[e]; }
    for (int e = 0; e < NE; ++e) { p1[e] /= sum; p1[e] = fminf(fmaxf(p1[e], -30.f), 30.f); }
    // softmax #2
    mx = p1[0];
    for (int e = 1; e < NE; ++e) mx = fmaxf(mx, p1[e]);
    float pr[NE]; sum = 0.f;
    for (int e = 0; e < NE; ++e) { pr[e] = __expf(p1[e] - mx); sum += pr[e]; }
    for (int e = 0; e < NE; ++e) pr[e] /= sum;
    // top-2
    int i0 = 0;
    for (int e = 1; e < NE; ++e) if (pr[e] > pr[i0]) i0 = e;
    int i1 = (i0 == 0) ? 1 : 0;
    for (int e = 0; e < NE; ++e) if (e != i0 && pr[e] > pr[i1]) i1 = e;
    float s2 = pr[i0] + pr[i1] + 1e-6f;
    if (lane == 0) {
        idx[b * KT + 0] = i0;          idx[b * KT + 1] = i1;
        rwt[b * KT + 0] = pr[i0] / s2; rwt[b * KT + 1] = pr[i1] / s2;
    }
}

// ---------------- kernel 3: shared 1x1 grouped conv via WMMA (+GN stats) ----------------
// per block: one (b, g), 4 x 64-pixel tiles.  GEMM 32(M) x 32(K) x 256(N), 8 waves.
__global__ void k_shared_conv(const float* __restrict__ x, const float* __restrict__ sw,
                              float* __restrict__ sconv, float* __restrict__ sstat) {
    __shared__ __align__(16) ushort_t lA[32 * 32];   // [m][k]
    __shared__ __align__(16) ushort_t lB[64 * 40];   // [px][ci], 80B rows (16B aligned)
    int bid = blockIdx.x;
    int sub = bid & 15, g = (bid >> 4) & 7, b = bid >> 7;
    for (int i = threadIdx.x; i < 1024; i += 256)
        lA[i] = f2bf(sw[(g * CG + (i >> 5)) * CG + (i & 31)]);
    int lane = threadIdx.x & 31, w = threadIdx.x >> 5;
    int mt = w & 1, nt = w >> 1, hi = lane >> 4, lm = lane & 15;
    const uint4* pa = (const uint4*)&lA[(mt * 16 + lm) * 32];
    const uint4* pb = (const uint4*)&lB[(nt * 16 + lm) * 40];
    float s = 0.f, q = 0.f;
    for (int t = 0; t < 4; ++t) {
        int p0 = sub * 256 + t * 64;
        for (int i = threadIdx.x; i < 2048; i += 256) {
            int ci = i >> 6, px = i & 63;
            lB[px * 40 + ci] = f2bf(x[((size_t)b * CIN + g * CG + ci) * HWP + p0 + px]);
        }
        __syncthreads();
        Frag A, Bf;
        A.q[0]  = pa[hi];      A.q[1]  = pa[2 + hi];
        Bf.q[0] = pb[hi];      Bf.q[1] = pb[2 + hi];
        v8f c = {0.f, 0.f, 0.f, 0.f, 0.f, 0.f, 0.f, 0.f};
        c = __builtin_amdgcn_wmma_f32_16x16x32_bf16(false, A.v, false, Bf.v, (short)0, c, false, false);
#pragma unroll
        for (int r = 0; r < 8; ++r) {
            int co = g * CG + mt * 16 + hi * 8 + r;
            float v = c[r];
            sconv[((size_t)b * COUT + co) * HWP + p0 + nt * 16 + lm] = v;
            s += v; q += v * v;
        }
        __syncthreads();
    }
    for (int o = 16; o > 0; o >>= 1) { s += __shfl_xor(s, o, 32); q += __shfl_xor(q, o, 32); }
    if (lane == 0) {
        atomicAdd(&sstat[(b * NG + g) * 2 + 0], s);
        atomicAdd(&sstat[(b * NG + g) * 2 + 1], q);
    }
}

// ---------------- kernel 4: expert 3x3 grouped conv via WMMA (+GN stats) ----------------
// per block: one (b,k), 128-channel M block, one 64-pixel image row (N=64).
// 8 waves: wave w owns channels [w*16, w*16+16) x all 64 px -> 4 accumulators.
// K = 288 in 9 chunks of 32; A fragment shared across the 4 WMMAs per chunk.
__global__ void k_expert_conv(const float* __restrict__ x, const float* __restrict__ fw,
                              const int* __restrict__ idx,
                              float* __restrict__ econv, float* __restrict__ estat) {
    __shared__ __align__(16) ushort_t lA[128 * 32];  // [m][k]  8 KB
    __shared__ __align__(16) ushort_t lB[64 * 40];   // [n][k]  5 KB, 80B rows
    int bid = blockIdx.x;
    int row = bid & 63, mblk = (bid >> 6) & 1, bk = bid >> 7;
    int b = bk >> 1, kk = bk & 1;
    int e = idx[b * KT + kk];
    int p0 = row * 64;                               // tile is one full image row
    const float* xc = x + ((size_t)b * CIN + e * CG) * HWP;    // conv-group == expert
    const float* wr = fw + (size_t)(e * COUT + mblk * 128) * KDIM;
    int lane = threadIdx.x & 31, w = threadIdx.x >> 5;
    int hi = lane >> 4, lm = lane & 15;
    const uint4* pa = (const uint4*)&lA[(w * 16 + lm) * 32];
    v8f acc[4];
#pragma unroll
    for (int j = 0; j < 4; ++j)
#pragma unroll
        for (int r = 0; r < 8; ++r) acc[j][r] = 0.f;
    for (int kc = 0; kc < 9; ++kc) {
        for (int i = threadIdx.x; i < 4096; i += 256)
            lA[i] = f2bf(wr[(size_t)(i >> 5) * KDIM + kc * 32 + (i & 31)]);
        if (kc < 8)   // global_prefetch_b8: pull next chunk's weight rows toward L2/L0
            __builtin_prefetch(wr + (size_t)(threadIdx.x >> 1) * KDIM + (kc + 1) * 32, 0, 1);
        for (int i = threadIdx.x; i < 2048; i += 256) {
            int n = i >> 5, kl = i & 31;
            int kkg = kc * 32 + kl;
            int ci = kkg / 9, r = kkg - ci * 9;
            int kh = r / 3, kw = r - kh * 3;
            int sy = row + kh - 1, sx = n + kw - 1;
            float v = (sy >= 0 && sy < HH && sx >= 0 && sx < WW)
                          ? xc[(size_t)ci * HWP + sy * WW + sx] : 0.f;
            lB[n * 40 + kl] = f2bf(v);
        }
        __syncthreads();
        Frag A;
        A.q[0] = pa[hi]; A.q[1] = pa[2 + hi];
#pragma unroll
        for (int j = 0; j < 4; ++j) {
            Frag Bf;
            const uint4* pb = (const uint4*)&lB[(j * 16 + lm) * 40];
            Bf.q[0] = pb[hi]; Bf.q[1] = pb[2 + hi];
            acc[j] = __builtin_amdgcn_wmma_f32_16x16x32_bf16(
                         false, A.v, false, Bf.v, (short)0, acc[j], false, false);
        }
        __syncthreads();
    }
    float s = 0.f, q = 0.f;
    int cobase = mblk * 128 + w * 16 + hi * 8;
    size_t obase = ((size_t)(b * KT + kk) * COUT + cobase) * HWP + p0 + lm;
#pragma unroll
    for (int j = 0; j < 4; ++j)
#pragma unroll
        for (int r = 0; r < 8; ++r) {
            float v = acc[j][r];
            econv[obase + (size_t)r * HWP + j * 16] = v;
            s += v; q += v * v;
        }
    for (int o = 16; o > 0; o >>= 1) { s += __shfl_xor(s, o, 32); q += __shfl_xor(q, o, 32); }
    int g2 = (mblk * 128 + w * 16) >> 5;    // GN group of this wave's 16 channels
    if (lane == 0) {
        atomicAdd(&estat[((b * KT + kk) * NG + g2) * 2 + 0], s);
        atomicAdd(&estat[((b * KT + kk) * NG + g2) * 2 + 1], q);
    }
}

// ---------------- kernel 5: GroupNorm + SiLU + weighted combine ----------------
__global__ void k_combine(const float* __restrict__ sconv, const float* __restrict__ econv,
                          const float* __restrict__ sstat, const float* __restrict__ estat,
                          const int* __restrict__ idx, const float* __restrict__ rwt,
                          const float* __restrict__ gsg, const float* __restrict__ gsb,
                          const float* __restrict__ eg, const float* __restrict__ eb,
                          float* __restrict__ out) {
    int bid = blockIdx.x;
    int co = bid & 255, b = bid >> 8;
    int g = co >> 5;
    const float N = (float)(CG * HWP);
    float ss = sstat[(b * NG + g) * 2], sq = sstat[(b * NG + g) * 2 + 1];
    float m_s = ss / N, v_s = fmaxf(sq / N - m_s * m_s, 0.f);
    float r_s = rsqrtf(v_s + 1e-5f);
    float ga_s = gsg[co], be_s = gsb[co];
    int e0 = idx[b * KT], e1 = idx[b * KT + 1];
    float w0 = rwt[b * KT], w1 = rwt[b * KT + 1];
    float s0 = estat[((b * KT + 0) * NG + g) * 2], q0 = estat[((b * KT + 0) * NG + g) * 2 + 1];
    float m0 = s0 / N, r0 = rsqrtf(fmaxf(q0 / N - m0 * m0, 0.f) + 1e-5f);
    float ga0 = eg[e0 * COUT + co], be0 = eb[e0 * COUT + co];
    float s1 = estat[((b * KT + 1) * NG + g) * 2], q1 = estat[((b * KT + 1) * NG + g) * 2 + 1];
    float m1 = s1 / N, r1 = rsqrtf(fmaxf(q1 / N - m1 * m1, 0.f) + 1e-5f);
    float ga1 = eg[e1 * COUT + co], be1 = eb[e1 * COUT + co];
    size_t base = ((size_t)b * COUT + co) * HWP;
    size_t eb0  = ((size_t)(b * KT + 0) * COUT + co) * HWP;
    size_t eb1  = ((size_t)(b * KT + 1) * COUT + co) * HWP;
    for (int p = threadIdx.x; p < HWP; p += 256) {
        float vs = sconv[base + p]; vs = silu((vs - m_s) * r_s * ga_s + be_s);
        float v0 = econv[eb0 + p];  v0 = silu((v0 - m0) * r0 * ga0 + be0);
        float v1 = econv[eb1 + p];  v1 = silu((v1 - m1) * r1 * ga1 + be1);
        out[base + p] = vs + w0 * v0 + w1 * v1;
    }
}

// ---------------- workspace layout (bytes) ----------------
#define OFF_MEAN   0u
#define OFF_STD    16384u
#define OFF_IDX    32768u
#define OFF_RW     32896u
#define OFF_SST    33280u                      // 16*8*2 floats  (1024 B)
#define OFF_EST    34304u                      // 16*2*8*2 floats (2048 B)
#define OFF_SCONV  36864u                      // 64 MB
#define OFF_ECONV  (36864u + 67108864u)        // 128 MB

extern "C" void kernel_launch(void* const* d_in, const int* in_sizes, int n_in,
                              void* d_out, int out_size, void* d_ws, size_t ws_size,
                              hipStream_t stream) {
    const float* x        = (const float*)d_in[0];
    const float* router_w = (const float*)d_in[1];
    const float* fused_w  = (const float*)d_in[2];
    const float* shared_w = (const float*)d_in[3];
    const float* gsg      = (const float*)d_in[4];
    const float* gsb      = (const float*)d_in[5];
    const float* eg       = (const float*)d_in[6];
    const float* ebp      = (const float*)d_in[7];
    float* out = (float*)d_out;
    char*  ws  = (char*)d_ws;

    float* mean  = (float*)(ws + OFF_MEAN);
    float* stdv  = (float*)(ws + OFF_STD);
    int*   idx   = (int*)  (ws + OFF_IDX);
    float* rwt   = (float*)(ws + OFF_RW);
    float* sstat = (float*)(ws + OFF_SST);
    float* estat = (float*)(ws + OFF_EST);
    float* sconv = (float*)(ws + OFF_SCONV);
    float* econv = (float*)(ws + OFF_ECONV);

    hipMemsetAsync(ws + OFF_SST, 0, 1024 + 2048, stream);   // zero GN stat accumulators

    k_chanstats  <<<NB * CIN,       256, 0, stream>>>(x, mean, stdv);
    k_router     <<<1,              512, 0, stream>>>(mean, stdv, router_w, idx, rwt);
    k_shared_conv<<<NB * NG * 16,   256, 0, stream>>>(x, shared_w, sconv, sstat);
    k_expert_conv<<<NB * KT * 2 * 64, 256, 0, stream>>>(x, fused_w, idx, econv, estat);
    k_combine    <<<NB * COUT,      256, 0, stream>>>(sconv, econv, sstat, estat,
                                                      idx, rwt, gsg, gsb, eg, ebp, out);
}
